// TemporalEigenstateNetwork_33114197852601
// MI455X (gfx1250) — matrix operand order repaired and checked by reading, not compiled, not run on hardware
//
#include <hip/hip_runtime.h>

// ---------------- problem constants ----------------
#define Bb 2
#define Tt 1024
#define Dd 512
#define Vv 32000
#define Ll 6
#define Cc 4
#define Kk 16
#define Ff 1024
#define Mm (Bb*Tt)          // 2048 rows of activations

typedef unsigned short u16;
typedef __attribute__((ext_vector_type(16))) __bf16 v16bf;
typedef __attribute__((ext_vector_type(8)))  __bf16 v8bf;
typedef __attribute__((ext_vector_type(8)))  float  v8f;

// ---------------- helpers ----------------
__device__ __forceinline__ u16 bf16rne(float f) {
    unsigned int u = __float_as_uint(f);
    unsigned int r = u + 0x7FFFu + ((u >> 16) & 1u);
    return (u16)(r >> 16);
}

// gfx1250 async copy: global -> LDS, 16 bytes per lane, tracked by ASYNCcnt.
__device__ __forceinline__ void async_ld_b128(unsigned lds_byte_addr, const void* gaddr) {
    asm volatile("global_load_async_to_lds_b128 %0, %1, off"
                 :: "v"(lds_byte_addr), "v"(gaddr)
                 : "memory");
}
__device__ __forceinline__ void wait_async_le2() { asm volatile("s_wait_asynccnt 0x2" ::: "memory"); }
__device__ __forceinline__ void wait_async_0()  { asm volatile("s_wait_asynccnt 0x0" ::: "memory"); }

// =====================================================================
// Generic NT GEMM:  C[m,n] = act( sum_k A[m,k]*B[n,k] + bias[n] )
// A: (M,Kd) bf16 row-major (lda), B: (N,Kd) bf16 row-major (ldb),
// C: (M,N) f32 (ldc). gridDim.z batches via element strides.
//
// Block = 256 threads = 8 waves computing a 128x64 C tile.
// B tiles (64 rows x 32 k) staged in LDS via global_load_async_to_lds_b128,
// two tiles (64 k) per sync epoch, 4-buffer pipeline, ASYNCcnt + barriers.
// A fragments streamed from global (no cross-wave A reuse exists).
// Requires M % 128 == 0, N % 64 == 0, Kd % 64 == 0 or Kd == 32.
// =====================================================================
#define BROW_PAD 40          // LDS row pitch in u16 elems (80 B): conflict-free, 16B aligned
#define BTILE_E  (64 * BROW_PAD)

__global__ __launch_bounds__(256)
void gemm_bf16_nt(const u16* __restrict__ A, int lda, long sAz,
                  const u16* __restrict__ B, int ldb, long sBz,
                  float* __restrict__ C, int ldc, long sCz,
                  const float* __restrict__ bias, long sBiasz,
                  int M, int N, int Kd, int act)
{
    const int z = blockIdx.z;
    const u16* Az = A + (long)z * sAz;
    const u16* Bz = B + (long)z * sBz;
    float*     Cz = C + (long)z * sCz;
    const float* bias_ = bias ? (bias + (long)z * sBiasz) : nullptr;

    const int mBlocks = M >> 7;                 // 128-row blocks
    const int mB = blockIdx.x % mBlocks;
    const int nG = blockIdx.x / mBlocks;
    const int waveIn = threadIdx.x >> 5;
    const int lane   = threadIdx.x & 31;
    const int m0 = (mB << 7) + (waveIn << 4);
    const int n0 = nG << 6;

    const int lhalf = lane >> 4;     // 0 or 1
    const int l16   = lane & 15;
    const int kOffA = lhalf << 4;    // A layout: K 0..15 (lanes 0-15), 16..31 (lanes 16-31), matching B
    const int kOffB = lhalf << 4;    // B layout (same order as A): K 0..15 (lanes 0-15), 16..31 (lanes 16-31)

    __shared__ u16 shB[4 * BTILE_E];            // 4 x 5120 B (two stages x two 32-k tiles)

    // staging map: 256 threads x 16 B cover one 64x32 bf16 tile
    const int srow   = threadIdx.x >> 2;        // 0..63
    const int schunk = threadIdx.x & 3;         // 0..3  (16 B chunks of a 64 B row)
    const unsigned ldsBase = (unsigned)(size_t)&shB[0];
    const unsigned stOff   = (unsigned)(srow * (BROW_PAD * 2) + schunk * 16);
    const u16* gB = Bz + (long)(n0 + srow) * ldb + schunk * 8;

    const u16* aRow = Az + (long)(m0 + l16) * lda;

    v8f acc0 = {}, acc1 = {}, acc2 = {}, acc3 = {};

    union AU { v16bf v; v8bf h[2]; };

    auto bufAddr = [&](int j) -> unsigned { return ldsBase + (unsigned)(j * (BTILE_E * 2)) + stOff; };

    auto computeChunk = [&](int buf, int k) {
        AU ua;
        ua.h[0] = *(const v8bf*)(aRow + k + kOffA);
        ua.h[1] = *(const v8bf*)(aRow + k + kOffA + 8);
        v16bf a = ua.v;
        __builtin_prefetch(aRow + k + 64, 0, 1);

        const u16* sb = &shB[buf * BTILE_E + l16 * BROW_PAD + kOffB];
        v16bf b0 = *(const v16bf*)(sb);
        v16bf b1 = *(const v16bf*)(sb + 16 * BROW_PAD);
        v16bf b2 = *(const v16bf*)(sb + 32 * BROW_PAD);
        v16bf b3 = *(const v16bf*)(sb + 48 * BROW_PAD);

        acc0 = __builtin_amdgcn_wmma_f32_16x16x32_bf16(false, a, false, b0, (short)0, acc0, false, false);
        acc1 = __builtin_amdgcn_wmma_f32_16x16x32_bf16(false, a, false, b1, (short)0, acc1, false, false);
        acc2 = __builtin_amdgcn_wmma_f32_16x16x32_bf16(false, a, false, b2, (short)0, acc2, false, false);
        acc3 = __builtin_amdgcn_wmma_f32_16x16x32_bf16(false, a, false, b3, (short)0, acc3, false, false);
    };

    if (Kd >= 64) {
        const int ns = Kd >> 6;                 // 64-k stages (Kd % 64 == 0)
        // prologue: stage 0 -> buffers 0,1
        async_ld_b128(bufAddr(0), gB);
        async_ld_b128(bufAddr(1), gB + 32);
        for (int s = 0; s < ns - 1; ++s) {
            const int p = s & 1;
            const int q = p ^ 1;
            const u16* gNext = gB + ((long)(s + 1) << 6);
            async_ld_b128(bufAddr(2 * q),     gNext);
            async_ld_b128(bufAddr(2 * q + 1), gNext + 32);
            wait_async_le2();                   // stage s's two copies done (in-order)
            __syncthreads();                    // visible to all waves
            computeChunk(2 * p,     (s << 6));
            computeChunk(2 * p + 1, (s << 6) + 32);
            __syncthreads();                    // reads done before buffers are reused
        }
        wait_async_0();
        __syncthreads();
        const int sl = ns - 1, p = sl & 1;
        computeChunk(2 * p,     (sl << 6));
        computeChunk(2 * p + 1, (sl << 6) + 32);
    } else {
        // Kd == 32 (cell output projection)
        async_ld_b128(bufAddr(0), gB);
        wait_async_0();
        __syncthreads();
        computeChunk(0, 0);
    }

    // Epilogue. C layout: VGPR r -> row m0+r (lanes 0-15) / m0+8+r (lanes 16-31), col = n-sub + l16.
    const int rBase = m0 + (lhalf << 3);
#define EPI(ACC, S)                                                          \
    {                                                                        \
        const int col = n0 + ((S) << 4) + l16;                               \
        const float bv = bias_ ? bias_[col] : 0.0f;                          \
        _Pragma("unroll")                                                    \
        for (int r = 0; r < 8; ++r) {                                        \
            float v = ACC[r] + bv;                                           \
            if (act == 1) v = v * (1.0f / (1.0f + __expf(-v)));              \
            Cz[(long)(rBase + r) * ldc + col] = v;                           \
        }                                                                    \
    }
    EPI(acc0, 0) EPI(acc1, 1) EPI(acc2, 2) EPI(acc3, 3)
#undef EPI
}

// =====================================================================
// Embedding: x[m,d] = token_emb[tok[m],d] + pos_emb[t,d]
// =====================================================================
__global__ void k_embed(const int* __restrict__ tok, const float* __restrict__ temb,
                        const float* __restrict__ pemb, float* __restrict__ x)
{
    long i = (long)blockIdx.x * blockDim.x + threadIdx.x;
    if (i >= (long)Mm * Dd) return;
    int m = (int)(i >> 9);
    int d = (int)(i & 511);
    int t = m & (Tt - 1);
    x[i] = temb[(long)tok[m] * Dd + d] + pemb[(long)t * Dd + d];
}

// f32 -> bf16 (grid-stride)
__global__ void k_f32_to_bf16(const float* __restrict__ in, u16* __restrict__ out, long n)
{
    long i = (long)blockIdx.x * blockDim.x + threadIdx.x;
    long s = (long)gridDim.x * blockDim.x;
    for (; i < n; i += s) out[i] = bf16rne(in[i]);
}

// reals (2048,64) -> bf16 padded (2048, C*32): col c*32+k, zeros for k>=16
__global__ void k_pack_reals(const float* __restrict__ r, u16* __restrict__ o)
{
    long i = (long)blockIdx.x * blockDim.x + threadIdx.x;
    if (i >= (long)Mm * Cc * 32) return;
    int k = (int)(i & 31);
    int c = (int)((i >> 5) & 3);
    int m = (int)(i >> 7);
    o[i] = (k < Kk) ? bf16rne(r[(long)m * (Cc * Kk) + c * Kk + k]) : (u16)0;
}

// wout (C,D,16) -> bf16 padded (C,D,32)
__global__ void k_pack_wout(const float* __restrict__ w, u16* __restrict__ o)
{
    long i = (long)blockIdx.x * blockDim.x + threadIdx.x;
    if (i >= (long)Cc * Dd * 32) return;
    int k = (int)(i & 31);
    long cd = i >> 5;
    o[i] = (k < Kk) ? bf16rne(w[cd * Kk + k]) : (u16)0;
}

// =====================================================================
// TemporalFlowCell scan: 1 wave per cell, lane = b*16 + k (B=2, K=16).
// Resonance column kept in VGPRs; nr/ni broadcast via LDS each step.
// =====================================================================
__global__ void k_scan(const float* __restrict__ alpha_l, const float* __restrict__ omega_l,
                       const float* __restrict__ res_l, const float* __restrict__ beta,
                       float* __restrict__ reals)
{
    const int c    = blockIdx.x;
    const int lane = threadIdx.x;
    const int b    = lane >> 4;
    const int kk   = lane & 15;

    const float a   = alpha_l[c * Kk + kk];
    const float mag = 1.0f / (1.0f + __expf(-a));
    const float om  = omega_l[c * Kk + kk];
    const float cp  = __cosf(om);
    const float sp  = __sinf(om);

    float rc[16];
#pragma unroll
    for (int k2 = 0; k2 < 16; ++k2) rc[k2] = res_l[c * 256 + k2 * 16 + kk];  // res[k2][kk]

    __shared__ float sh[64];
    float sr = 0.0f, si = 0.0f;
    const int base = b << 4;

    for (int t = 0; t < Tt; ++t) {
        long idx = ((long)(b * Tt + t)) * (Cc * Kk) + c * Kk + kk;
        float bt = beta[idx];
        float nr = mag * (sr * cp - si * sp) + bt;
        float ni = mag * (sr * sp + si * cp);
        sh[lane]      = nr;
        sh[32 + lane] = ni;
        __syncthreads();
        float ar = 0.0f, ai = 0.0f;
#pragma unroll
        for (int k2 = 0; k2 < 16; ++k2) {
            ar += sh[base + k2]      * rc[k2];
            ai += sh[32 + base + k2] * rc[k2];
        }
        __syncthreads();
        sr = ar; si = ai;
        reals[idx] = ar;
    }
}

// ---------------- wave-level LayerNorm helpers (32 threads/row, D=512) ----------------
__device__ __forceinline__ void wave_stats(float s, float q, float& mean, float& inv)
{
#pragma unroll
    for (int off = 16; off > 0; off >>= 1) {
        s += __shfl_xor(s, off, 32);
        q += __shfl_xor(q, off, 32);
    }
    mean = s * (1.0f / Dd);
    float var = q * (1.0f / Dd) - mean * mean;
    inv = rsqrtf(var + 1e-5f);
}

// Per-cell LN over Y (C*2048 rows), in place. bout already added as GEMM bias.
__global__ void k_cell_ln(float* __restrict__ Y, const float* __restrict__ w, const float* __restrict__ b)
{
    int row = blockIdx.x;
    int c = row >> 11;
    int lane = threadIdx.x;
    float* yr = Y + (long)row * Dd;
    const float* wc = w + c * Dd;
    const float* bc = b + c * Dd;
    float v[16]; float s = 0, q = 0;
#pragma unroll
    for (int i = 0; i < 16; ++i) { float t = yr[lane + 32 * i]; v[i] = t; s += t; q += t * t; }
    float mean, inv; wave_stats(s, q, mean, inv);
#pragma unroll
    for (int i = 0; i < 16; ++i) { int d = lane + 32 * i; yr[d] = (v[i] - mean) * inv * wc[d] + bc[d]; }
}

// x = LN(x + mean_c Y[c])  (n1)
__global__ void k_mix_ln(float* __restrict__ x, const float* __restrict__ Y,
                         const float* __restrict__ w, const float* __restrict__ b)
{
    int m = blockIdx.x;
    int lane = threadIdx.x;
    float* xr = x + (long)m * Dd;
    float v[16]; float s = 0, q = 0;
#pragma unroll
    for (int i = 0; i < 16; ++i) {
        int d = lane + 32 * i;
        float acc = xr[d];
        float ym = 0.0f;
#pragma unroll
        for (int c = 0; c < Cc; ++c) ym += Y[((long)c * Mm + m) * Dd + d];
        float t = acc + 0.25f * ym;
        v[i] = t; s += t; q += t * t;
    }
    float mean, inv; wave_stats(s, q, mean, inv);
#pragma unroll
    for (int i = 0; i < 16; ++i) { int d = lane + 32 * i; xr[d] = (v[i] - mean) * inv * w[d] + b[d]; }
}

// x = LN(x + ffn)  (n2)
__global__ void k_res_ln(float* __restrict__ x, const float* __restrict__ f,
                         const float* __restrict__ w, const float* __restrict__ b)
{
    int m = blockIdx.x;
    int lane = threadIdx.x;
    float* xr = x + (long)m * Dd;
    const float* fr = f + (long)m * Dd;
    float v[16]; float s = 0, q = 0;
#pragma unroll
    for (int i = 0; i < 16; ++i) {
        int d = lane + 32 * i;
        float t = xr[d] + fr[d];
        v[i] = t; s += t; q += t * t;
    }
    float mean, inv; wave_stats(s, q, mean, inv);
#pragma unroll
    for (int i = 0; i < 16; ++i) { int d = lane + 32 * i; xr[d] = (v[i] - mean) * inv * w[d] + b[d]; }
}

// xln_bf = bf16(LN(x))  (final norm)
__global__ void k_final_ln(const float* __restrict__ x, u16* __restrict__ o,
                           const float* __restrict__ w, const float* __restrict__ b)
{
    int m = blockIdx.x;
    int lane = threadIdx.x;
    const float* xr = x + (long)m * Dd;
    float v[16]; float s = 0, q = 0;
#pragma unroll
    for (int i = 0; i < 16; ++i) { float t = xr[lane + 32 * i]; v[i] = t; s += t; q += t * t; }
    float mean, inv; wave_stats(s, q, mean, inv);
#pragma unroll
    for (int i = 0; i < 16; ++i) {
        int d = lane + 32 * i;
        o[(long)m * Dd + d] = bf16rne((v[i] - mean) * inv * w[d] + b[d]);
    }
}

// h_bf = bf16(gate * up)
__global__ void k_gateup(const float* __restrict__ g, const float* __restrict__ u,
                         u16* __restrict__ h, long n)
{
    long i = (long)blockIdx.x * blockDim.x + threadIdx.x;
    long s = (long)gridDim.x * blockDim.x;
    for (; i < n; i += s) h[i] = bf16rne(g[i] * u[i]);
}

// =====================================================================
// Host orchestration
// =====================================================================
static inline size_t alignup(size_t x) { return (x + 255) & ~(size_t)255; }

extern "C" void kernel_launch(void* const* d_in, const int* in_sizes, int n_in,
                              void* d_out, int out_size, void* d_ws, size_t ws_size,
                              hipStream_t stream)
{
    (void)in_sizes; (void)n_in; (void)out_size; (void)ws_size;

    const int*   tokens = (const int*)  d_in[0];
    const float* temb   = (const float*)d_in[1];
    const float* pemb   = (const float*)d_in[2];
    const float* alpha  = (const float*)d_in[3];
    const float* omega  = (const float*)d_in[4];
    const float* win    = (const float*)d_in[5];
    const float* res    = (const float*)d_in[6];
    const float* wout   = (const float*)d_in[7];
    const float* bout   = (const float*)d_in[8];
    const float* cnw    = (const float*)d_in[9];
    const float* cnb    = (const float*)d_in[10];
    const float* gw     = (const float*)d_in[11];
    const float* gb     = (const float*)d_in[12];
    const float* uw     = (const float*)d_in[13];
    const float* ub     = (const float*)d_in[14];
    const float* dw     = (const float*)d_in[15];
    const float* db     = (const float*)d_in[16];
    const float* n1w    = (const float*)d_in[17];
    const float* n1b    = (const float*)d_in[18];
    const float* n2w    = (const float*)d_in[19];
    const float* n2b    = (const float*)d_in[20];
    const float* fnw    = (const float*)d_in[21];
    const float* fnb    = (const float*)d_in[22];
    float* out = (float*)d_out;

    // ---- workspace carve-out ----
    char* ws = (char*)d_ws;
    size_t off = 0;
    auto carve = [&](size_t bytes) { void* p = ws + off; off += alignup(bytes); return p; };

    u16*   temb_bf   = (u16*)  carve((size_t)Vv * Dd * 2);       // 32.8 MB
    float* x         = (float*)carve((size_t)Mm * Dd * 4);
    u16*   x_bf      = (u16*)  carve((size_t)Mm * Dd * 2);
    float* beta      = (float*)carve((size_t)Mm * Cc * Kk * 4);
    float* reals     = (float*)carve((size_t)Mm * Cc * Kk * 4);
    u16*   reals_bf  = (u16*)  carve((size_t)Mm * Cc * 32 * 2);
    float* Y         = (float*)carve((size_t)Cc * Mm * Dd * 4);  // 16.8 MB
    float* gatebuf   = (float*)carve((size_t)Mm * Ff * 4);
    float* upbuf     = (float*)carve((size_t)Mm * Ff * 4);
    u16*   h_bf      = (u16*)  carve((size_t)Mm * Ff * 2);
    float* ffnbuf    = (float*)carve((size_t)Mm * Dd * 4);
    u16*   xln_bf    = (u16*)  carve((size_t)Mm * Dd * 2);
    u16*   wincat_bf = (u16*)  carve((size_t)Cc * Kk * Dd * 2);
    u16*   wout_bf   = (u16*)  carve((size_t)Cc * Dd * 32 * 2);
    u16*   gw_bf     = (u16*)  carve((size_t)Ff * Dd * 2);
    u16*   uw_bf     = (u16*)  carve((size_t)Ff * Dd * 2);
    u16*   dw_bf     = (u16*)  carve((size_t)Dd * Ff * 2);

    auto gemm = [&](const u16* A, int lda, long sAz, const u16* Bp, int ldb, long sBz,
                    float* Cp, int ldc, long sCz, const float* bias, long sbz,
                    int M, int N, int Kd, int act, int zDim) {
        int mBlocks = M >> 7;                 // M % 128 == 0 for all calls
        int nGroups = N >> 6;
        dim3 grid(mBlocks * nGroups, 1, zDim);
        gemm_bf16_nt<<<grid, 256, 0, stream>>>(A, lda, sAz, Bp, ldb, sBz,
                                               Cp, ldc, sCz, bias, sbz, M, N, Kd, act);
    };
    auto conv = [&](const float* in, u16* o, long n) {
        int blocks = (int)((n + 1023) / 1024); if (blocks > 4096) blocks = 4096;
        k_f32_to_bf16<<<blocks, 256, 0, stream>>>(in, o, n);
    };

    // ---- once: embedding + bf16 token embedding ----
    k_embed<<<(Mm * Dd + 255) / 256, 256, 0, stream>>>(tokens, temb, pemb, x);
    conv(temb, temb_bf, (long)Vv * Dd);

    for (int l = 0; l < Ll; ++l) {
        const float* alpha_l = alpha + (long)l * Cc * Kk;
        const float* omega_l = omega + (long)l * Cc * Kk;
        const float* win_l   = win   + (long)l * Cc * Kk * Dd;
        const float* res_l   = res   + (long)l * Cc * Kk * Kk;
        const float* wout_l  = wout  + (long)l * Cc * Dd * Kk;
        const float* bout_l  = bout  + (long)l * Cc * Dd;
        const float* cnw_l   = cnw   + (long)l * Cc * Dd;
        const float* cnb_l   = cnb   + (long)l * Cc * Dd;
        const float* gw_l    = gw + (long)l * Ff * Dd;
        const float* gb_l    = gb + (long)l * Ff;
        const float* uw_l    = uw + (long)l * Ff * Dd;
        const float* ub_l    = ub + (long)l * Ff;
        const float* dw_l    = dw + (long)l * Dd * Ff;
        const float* db_l    = db + (long)l * Dd;

        // ---- cells ----
        conv(x, x_bf, (long)Mm * Dd);
        conv(win_l, wincat_bf, (long)Cc * Kk * Dd);   // win[l] is (C*K, D): direct N=64 weight
        // beta = x @ wincat^T  : (2048, 64)
        gemm(x_bf, Dd, 0, wincat_bf, Dd, 0, beta, Cc * Kk, 0, nullptr, 0,
             Mm, Cc * Kk, Dd, 0, 1);
        // sequential resonance scan, 4 cells in parallel (one wave each)
        k_scan<<<Cc, 32, 0, stream>>>(alpha_l, omega_l, res_l, beta, reals);
        // per-cell output projection (K=16 zero-padded to 32), bias = bout[c]
        k_pack_reals<<<(Mm * Cc * 32 + 255) / 256, 256, 0, stream>>>(reals, reals_bf);
        k_pack_wout<<<(Cc * Dd * 32 + 255) / 256, 256, 0, stream>>>(wout_l, wout_bf);
        gemm(reals_bf, Cc * 32, /*sAz=*/32, wout_bf, 32, /*sBz=*/(long)Dd * 32,
             Y, Dd, /*sCz=*/(long)Mm * Dd, bout_l, /*sbz=*/Dd,
             Mm, Dd, 32, 0, Cc);
        k_cell_ln<<<Cc * Mm, 32, 0, stream>>>(Y, cnw_l, cnb_l);
        k_mix_ln<<<Mm, 32, 0, stream>>>(x, Y, n1w + (long)l * Dd, n1b + (long)l * Dd);

        // ---- FFN (SwiGLU) ----
        conv(x, x_bf, (long)Mm * Dd);
        conv(gw_l, gw_bf, (long)Ff * Dd);
        conv(uw_l, uw_bf, (long)Ff * Dd);
        conv(dw_l, dw_bf, (long)Dd * Ff);
        gemm(x_bf, Dd, 0, gw_bf, Dd, 0, gatebuf, Ff, 0, gb_l, 0, Mm, Ff, Dd, /*silu*/1, 1);
        gemm(x_bf, Dd, 0, uw_bf, Dd, 0, upbuf,   Ff, 0, ub_l, 0, Mm, Ff, Dd, 0, 1);
        k_gateup<<<4096, 256, 0, stream>>>(gatebuf, upbuf, h_bf, (long)Mm * Ff);
        gemm(h_bf, Ff, 0, dw_bf, Ff, 0, ffnbuf, Dd, 0, db_l, 0, Mm, Dd, Ff, 0, 1);
        k_res_ln<<<Mm, 32, 0, stream>>>(x, ffnbuf, n2w + (long)l * Dd, n2b + (long)l * Dd);
    }

    // ---- head: LN(x) @ token_emb^T -> (2048, 32000) f32 ----
    k_final_ln<<<Mm, 32, 0, stream>>>(x, xln_bf, fnw, fnb);
    gemm(xln_bf, Dd, 0, temb_bf, Dd, 0, out, Vv, 0, nullptr, 0, Mm, Vv, Dd, 0, 1);
}
